// NSMNN_66065186947122
// MI455X (gfx1250) — compile-verified, matrix operand order checked by reading, not run on hardware
//
#include <hip/hip_runtime.h>
#include <cstdint>
#include <cstddef>

// ---------------------------------------------------------------------------
// MI455X (gfx1250) MoE controller forward pass.
// All GEMMs on V_WMMA_F32_16X16X32_BF16 (f32 accumulate).  f32 operands are
// converted to bf16 while staging into double-buffered LDS; global fetches of
// tile k+1 are issued before the WMMAs of tile k (fetch-early / store-late
// software pipeline), bounds checks only in a uniform-branch tail path.
// ---------------------------------------------------------------------------

typedef __attribute__((ext_vector_type(16))) __bf16 v16bf;
typedef __attribute__((ext_vector_type(2)))  __bf16 bf16x2;
typedef __attribute__((ext_vector_type(8)))  float  v8f;

#define BATCH 32768
#define NEXP  10

constexpr int BM = 128;   // block tile M
constexpr int BN = 128;   // block tile N
constexpr int BK = 32;    // block tile K (== WMMA K)
constexpr int STR = 40;   // padded LDS k-stride -> conflict-free b128 frag loads
constexpr int NTHREADS = 256;  // 8 wave32 waves

struct U32x8 { uint4 lo, hi; };

// Load one 16x32 bf16 operand fragment from LDS (row-major, stride STR).
// CDNA5 16-bit A layout: lane<16 holds K {0..7,16..23} of row lane,
// lane>=16 holds K {8..15,24..31} of row lane-16.  Two ds_load_b128 per frag.
__device__ __forceinline__ v16bf lds_frag(const __bf16* base, int lane) {
  const int r = lane & 15;
  const int h = (lane >> 4) & 1;
  const __bf16* q = base + r * STR + h * 8;
  U32x8 t;
  t.lo = *(const uint4*)(q);
  t.hi = *(const uint4*)(q + 16);
  return __builtin_bit_cast(v16bf, t);
}

__device__ __forceinline__ float elu_f(float x) {
  return x > 0.f ? x : __expf(x) - 1.f;
}

// ---- staging helpers: 256 threads move a 128x32 (A) / 32x128 (B) f32 tile --
// fetch_*  : global -> registers (issued BEFORE the WMMAs of the current tile)
// store_*  : registers -> LDS bf16 (issued AFTER the WMMAs)

template<bool KCHK>
__device__ __forceinline__ void fetch_A16(const float* __restrict__ A, int lda,
                                          int m0, int k0, int K, int t, float* r) {
  #pragma unroll
  for (int it = 0; it < 8; ++it) {
    const int eid = it * 512 + t * 2;
    const int row = eid >> 5;          // 0..127
    const int col = eid & 31;          // even
    const float* ap = A + (size_t)(m0 + row) * lda + k0 + col;
    if constexpr (!KCHK) {
      const float2 f = *(const float2*)ap;
      r[2 * it] = f.x; r[2 * it + 1] = f.y;
    } else {
      const int kk = k0 + col;
      float v0 = 0.f, v1 = 0.f;
      if (kk + 1 < K)  { const float2 f = *(const float2*)ap; v0 = f.x; v1 = f.y; }
      else if (kk < K) { v0 = ap[0]; }
      r[2 * it] = v0; r[2 * it + 1] = v1;
    }
  }
}

__device__ __forceinline__ void store_A16(__bf16* As, int t, const float* r) {
  #pragma unroll
  for (int it = 0; it < 8; ++it) {
    const int eid = it * 512 + t * 2;
    const int row = eid >> 5, col = eid & 31;
    bf16x2 p; p.x = (__bf16)r[2 * it]; p.y = (__bf16)r[2 * it + 1];
    *(bf16x2*)&As[row * STR + col] = p;
  }
}

// MoE variant: scale row by its gate weight for expert e while converting.
__device__ __forceinline__ void store_A16_scaled(__bf16* As, int t, const float* r,
                                                 const float* Wg, int e) {
  #pragma unroll
  for (int it = 0; it < 8; ++it) {
    const int eid = it * 512 + t * 2;
    const int row = eid >> 5, col = eid & 31;
    const float s = Wg[row * NEXP + e];
    bf16x2 p; p.x = (__bf16)(r[2 * it] * s); p.y = (__bf16)(r[2 * it + 1] * s);
    *(bf16x2*)&As[row * STR + col] = p;
  }
}

template<bool KCHK, bool NCHK>
__device__ __forceinline__ void fetch_B16(const float* __restrict__ W, int ldb,
                                          int k0, int K, int n0, int N, int t, float* r) {
  #pragma unroll
  for (int it = 0; it < 8; ++it) {
    const int eid = it * 512 + t * 2;
    const int kr = eid >> 7;           // 0..31
    const int nc = eid & 127;          // even
    const int kk = k0 + kr, nn = n0 + nc;
    float v0 = 0.f, v1 = 0.f;
    if (!KCHK || kk < K) {
      const float* wp = W + (size_t)kk * ldb + nn;
      if (!NCHK || nn + 1 < N) { const float2 f = *(const float2*)wp; v0 = f.x; v1 = f.y; }
      else if (nn < N)         { v0 = wp[0]; }
    }
    r[2 * it] = v0; r[2 * it + 1] = v1;
  }
}

__device__ __forceinline__ void store_B16(__bf16* Bs, int t, const float* r) {
  #pragma unroll
  for (int it = 0; it < 8; ++it) {
    const int eid = it * 512 + t * 2;
    const int kr = eid >> 7, nc = eid & 127;
    Bs[(nc + 0) * STR + kr] = (__bf16)r[2 * it];
    Bs[(nc + 1) * STR + kr] = (__bf16)r[2 * it + 1];
  }
}

// ---------------------------------------------------------------------------
// Generic GEMM:  C[m,n] = act(sum_k A[m,k]*W[k,n] + bias[n])
// ---------------------------------------------------------------------------
__global__ __launch_bounds__(NTHREADS)
void gemm_bias_act(const float* __restrict__ A, int lda, int K,
                   const float* __restrict__ W, int ldb,
                   const float* __restrict__ bias,
                   float* __restrict__ C, int ldc, int N, int do_elu)
{
  __shared__ __align__(16) __bf16 As[2][BM * STR];
  __shared__ __align__(16) __bf16 Bs[2][BN * STR];

  const int t    = threadIdx.x;
  const int lane = t & 31;
  const int wid  = t >> 5;
  const int wm   = wid >> 1;           // 0..3 : 32 rows each
  const int wn   = wid & 1;            // 0..1 : 64 cols each
  const int m0   = blockIdx.y * BM;
  const int n0   = blockIdx.x * BN;
  const bool nfull  = (n0 + BN <= N);
  const int  ksteps = (K + BK - 1) / BK;

  float ra[16], rb[16];
  auto fetch = [&](int k0) {   // uniform-branch dispatch: checks only on tail
    if (k0 + BK <= K) {
      fetch_A16<false>(A, lda, m0, k0, K, t, ra);
      if (nfull) fetch_B16<false, false>(W, ldb, k0, K, n0, N, t, rb);
      else       fetch_B16<false, true >(W, ldb, k0, K, n0, N, t, rb);
    } else {
      fetch_A16<true >(A, lda, m0, k0, K, t, ra);
      if (nfull) fetch_B16<true , false>(W, ldb, k0, K, n0, N, t, rb);
      else       fetch_B16<true , true >(W, ldb, k0, K, n0, N, t, rb);
    }
  };

  const v8f vzero = {0.f,0.f,0.f,0.f,0.f,0.f,0.f,0.f};
  v8f acc[2][4];
  #pragma unroll
  for (int i = 0; i < 2; ++i)
    #pragma unroll
    for (int j = 0; j < 4; ++j) acc[i][j] = vzero;

  fetch(0);
  store_A16(As[0], t, ra);
  store_B16(Bs[0], t, rb);

  for (int ks = 0; ks < ksteps; ++ks) {
    __syncthreads();
    const int  cur  = ks & 1;
    const bool more = (ks + 1 < ksteps);
    if (more) fetch((ks + 1) * BK);     // global loads overlap the WMMAs below

    v16bf af[2], bfr[4];
    af[0] = lds_frag(&As[cur][(wm * 32 +  0) * STR], lane);
    af[1] = lds_frag(&As[cur][(wm * 32 + 16) * STR], lane);
    #pragma unroll
    for (int j = 0; j < 4; ++j)
      bfr[j] = lds_frag(&Bs[cur][(wn * 64 + j * 16) * STR], lane);

    #pragma unroll
    for (int i = 0; i < 2; ++i)
      #pragma unroll
      for (int j = 0; j < 4; ++j)
        acc[i][j] = __builtin_amdgcn_wmma_f32_16x16x32_bf16(
            false, af[i], false, bfr[j], (short)0, acc[i][j], false, false);

    if (more) {                         // convert + stage after the math
      store_A16(As[cur ^ 1], t, ra);
      store_B16(Bs[cur ^ 1], t, rb);
    }
  }

  // ---- epilogue: bias + activation + store (16 consecutive floats / half-wave)
  const int lh = (lane >> 4) & 1;
  const int nc = lane & 15;
  #pragma unroll
  for (int i = 0; i < 2; ++i) {
    #pragma unroll
    for (int j = 0; j < 4; ++j) {
      const int ng = n0 + wn * 64 + j * 16 + nc;
      if (ng >= N) continue;
      const float bn = bias[ng];
      #pragma unroll
      for (int r = 0; r < 8; ++r) {
        const int mg = m0 + wm * 32 + i * 16 + r + lh * 8;
        float v = acc[i][j][r] + bn;
        if (do_elu) v = elu_f(v);
        C[(size_t)mg * ldc + ng] = v;
      }
    }
  }
}

// ---------------------------------------------------------------------------
// Blended-expert GEMM:
//   C[m,n] = act( sum_e sum_k (gate[m,e]*X[m,k]) * W[e*K0+k, n]
//                 + sum_e gate[m,e]*biasE[e,n] )
// Gate scale rides on the A operand (one matrix row per lane).  K0 % 32 == 0,
// so every K-step stays inside one expert's block and needs no K checks.
// ---------------------------------------------------------------------------
__global__ __launch_bounds__(NTHREADS)
void moe_gemm(const float* __restrict__ X, int ldx, int K0,
              const float* __restrict__ W,      // (NEXP*K0, N) row-major
              const float* __restrict__ biasE,  // (NEXP, N)
              const float* __restrict__ gate,   // (BATCH, NEXP)
              float* __restrict__ C, int ldc, int N, int do_elu)
{
  __shared__ __align__(16) __bf16 As[2][BM * STR];
  __shared__ __align__(16) __bf16 Bs[2][BN * STR];
  __shared__ float Wg[BM * NEXP];

  const int t    = threadIdx.x;
  const int lane = t & 31;
  const int wid  = t >> 5;
  const int wm   = wid >> 1;
  const int wn   = wid & 1;
  const int m0   = blockIdx.y * BM;
  const int n0   = blockIdx.x * BN;
  const bool nfull = (n0 + BN <= N);
  const int  Ktot  = NEXP * K0;
  const int  ksteps = Ktot / BK;

  for (int idx = t; idx < BM * NEXP; idx += NTHREADS)
    Wg[idx] = gate[(size_t)m0 * NEXP + idx];    // contiguous block copy

  float ra[16], rb[16];
  auto fetch = [&](int k0) {
    const int e  = k0 / K0;
    const int kb = k0 - e * K0;
    fetch_A16<false>(X, ldx, m0, kb, Ktot, t, ra);
    if (nfull) fetch_B16<false, false>(W, N, k0, Ktot, n0, N, t, rb);
    else       fetch_B16<false, true >(W, N, k0, Ktot, n0, N, t, rb);
  };

  const v8f vzero = {0.f,0.f,0.f,0.f,0.f,0.f,0.f,0.f};
  v8f acc[2][4];
  #pragma unroll
  for (int i = 0; i < 2; ++i)
    #pragma unroll
    for (int j = 0; j < 4; ++j) acc[i][j] = vzero;

  fetch(0);
  __syncthreads();                      // Wg visible before first scaled store
  store_A16_scaled(As[0], t, ra, Wg, 0);
  store_B16(Bs[0], t, rb);

  for (int ks = 0; ks < ksteps; ++ks) {
    __syncthreads();
    const int  cur  = ks & 1;
    const bool more = (ks + 1 < ksteps);
    if (more) fetch((ks + 1) * BK);

    v16bf af[2], bfr[4];
    af[0] = lds_frag(&As[cur][(wm * 32 +  0) * STR], lane);
    af[1] = lds_frag(&As[cur][(wm * 32 + 16) * STR], lane);
    #pragma unroll
    for (int j = 0; j < 4; ++j)
      bfr[j] = lds_frag(&Bs[cur][(wn * 64 + j * 16) * STR], lane);

    #pragma unroll
    for (int i = 0; i < 2; ++i)
      #pragma unroll
      for (int j = 0; j < 4; ++j)
        acc[i][j] = __builtin_amdgcn_wmma_f32_16x16x32_bf16(
            false, af[i], false, bfr[j], (short)0, acc[i][j], false, false);

    if (more) {
      store_A16_scaled(As[cur ^ 1], t, ra, Wg, ((ks + 1) * BK) / K0);
      store_B16(Bs[cur ^ 1], t, rb);
    }
  }

  // ---- epilogue: blended expert bias + activation ------------------------
  const int lh = (lane >> 4) & 1;
  const int nc = lane & 15;
  #pragma unroll
  for (int i = 0; i < 2; ++i) {
    #pragma unroll
    for (int j = 0; j < 4; ++j) {
      const int ng = n0 + wn * 64 + j * 16 + nc;
      if (ng >= N) continue;
      float bn[NEXP];
      #pragma unroll
      for (int e = 0; e < NEXP; ++e) bn[e] = biasE[(size_t)e * N + ng];
      #pragma unroll
      for (int r = 0; r < 8; ++r) {
        const int ml = wm * 32 + i * 16 + r + lh * 8;   // block-local row
        float bb = 0.f;
        #pragma unroll
        for (int e = 0; e < NEXP; ++e) bb += Wg[ml * NEXP + e] * bn[e];
        float v = acc[i][j][r] + bb;
        if (do_elu) v = elu_f(v);
        C[(size_t)(m0 + ml) * ldc + ng] = v;
      }
    }
  }
}

// ---------------------------------------------------------------------------
// Gating head: out[row,:] = softmax(G[row,:] @ W2 + b2), one wave per row.
// ---------------------------------------------------------------------------
__global__ __launch_bounds__(256)
void gate_softmax(const float* __restrict__ G, const float* __restrict__ W2,
                  const float* __restrict__ b2, float* __restrict__ out)
{
  const int lane = threadIdx.x & 31;
  const int row  = blockIdx.x * 8 + (threadIdx.x >> 5);

  float acc[NEXP];
  #pragma unroll
  for (int e = 0; e < NEXP; ++e) acc[e] = 0.f;

  const float* g = G + (size_t)row * 512;
  for (int k = lane; k < 512; k += 32) {
    const float x = g[k];
    const float* w = W2 + (size_t)k * NEXP;
    #pragma unroll
    for (int e = 0; e < NEXP; ++e) acc[e] += x * w[e];
  }
  #pragma unroll
  for (int e = 0; e < NEXP; ++e) {
    #pragma unroll
    for (int off = 16; off > 0; off >>= 1)
      acc[e] += __shfl_xor(acc[e], off, 32);
    acc[e] += b2[e];
  }
  float mx = acc[0];
  #pragma unroll
  for (int e = 1; e < NEXP; ++e) mx = fmaxf(mx, acc[e]);
  float ex[NEXP], s = 0.f;
  #pragma unroll
  for (int e = 0; e < NEXP; ++e) { ex[e] = __expf(acc[e] - mx); s += ex[e]; }
  const float inv = 1.f / s;
  if (lane == 0) {
    #pragma unroll
    for (int e = 0; e < NEXP; ++e) out[(size_t)row * NEXP + e] = ex[e] * inv;
  }
}

// ---------------------------------------------------------------------------
extern "C" void kernel_launch(void* const* d_in, const int* in_sizes, int n_in,
                              void* d_out, int out_size, void* d_ws, size_t ws_size,
                              hipStream_t stream)
{
  const float* input_ = (const float*)d_in[0];
  const float* e0_W0  = (const float*)d_in[1];
  const float* e0_b0  = (const float*)d_in[2];
  const float* e0_W1  = (const float*)d_in[3];
  const float* e0_b1  = (const float*)d_in[4];
  const float* e1_W0  = (const float*)d_in[5];
  const float* e1_b0  = (const float*)d_in[6];
  const float* e1_W1  = (const float*)d_in[7];
  const float* e1_b1  = (const float*)d_in[8];
  const float* g_W0   = (const float*)d_in[9];
  const float* g_b0   = (const float*)d_in[10];
  const float* g_W1   = (const float*)d_in[11];
  const float* g_b1   = (const float*)d_in[12];
  const float* g_W2   = (const float*)d_in[13];
  const float* g_b2   = (const float*)d_in[14];
  const float* m_W0   = (const float*)d_in[15];
  const float* m_b0   = (const float*)d_in[16];
  const float* m_W1   = (const float*)d_in[17];
  const float* m_b1   = (const float*)d_in[18];
  const float* m_W2   = (const float*)d_in[19];
  const float* m_b2   = (const float*)d_in[20];

  float* out = (float*)d_out;

  // workspace (floats): feat(640) | t0(512) | g0(512) | g1(512) | t1(128) | wg(10)
  const size_t Bsz = BATCH;
  float* ws   = (float*)d_ws;
  float* feat = ws;                    // concat(h0,h1)
  float* t0   = feat + Bsz * 640;      // encoder0 hidden; later expert-layer-1 out
  float* g0   = t0   + Bsz * 512;      // gate hidden0;    later expert-layer-0 out
  float* g1   = g0   + Bsz * 512;      // gate hidden1
  float* t1   = g1   + Bsz * 512;      // encoder1 hidden
  float* wg   = t1   + Bsz * 128;      // softmax gate weights (B,10)

  dim3 blk(NTHREADS);
  dim3 grid512(512 / BN, BATCH / BM);
  dim3 grid128(128 / BN, BATCH / BM);
  dim3 grid578((578 + BN - 1) / BN, BATCH / BM);

  // encoder 0 (pose = input_[:,0:400])
  gemm_bias_act<<<grid512, blk, 0, stream>>>(input_,        556, 400, e0_W0, 512, e0_b0, t0,         512, 512, 1);
  gemm_bias_act<<<grid512, blk, 0, stream>>>(t0,            512, 512, e0_W1, 512, e0_b1, feat,       640, 512, 1);
  // encoder 1 (goal = input_[:,400:478]) -> feat[:,512:640]
  gemm_bias_act<<<grid128, blk, 0, stream>>>(input_ + 400,  556,  78, e1_W0, 128, e1_b0, t1,         128, 128, 1);
  gemm_bias_act<<<grid128, blk, 0, stream>>>(t1,            128, 128, e1_W1, 128, e1_b1, feat + 512, 640, 128, 1);
  // gating (gate_in = input_[:,478:556])
  gemm_bias_act<<<grid512, blk, 0, stream>>>(input_ + 478,  556,  78, g_W0,  512, g_b0,  g0,         512, 512, 1);
  gemm_bias_act<<<grid512, blk, 0, stream>>>(g0,            512, 512, g_W1,  512, g_b1,  g1,         512, 512, 1);
  gate_softmax<<<BATCH / 8, 256, 0, stream>>>(g1, g_W2, g_b2, wg);
  // blended experts
  moe_gemm<<<grid512, blk, 0, stream>>>(feat, 640, 640, m_W0, m_b0, wg, g0,  512, 512, 1);
  moe_gemm<<<grid512, blk, 0, stream>>>(g0,   512, 512, m_W1, m_b1, wg, t0,  512, 512, 1);
  moe_gemm<<<grid578, blk, 0, stream>>>(t0,   512, 512, m_W2, m_b2, wg, out, 578, 578, 0);
}
